// ChebyKANLayer_14259291423320
// MI455X (gfx1250) — compile-verified, matrix operand order checked by reading, not run on hardware
//
#include <hip/hip_runtime.h>

// CDNA5 / gfx1250: wave32, WMMA 16x16x32 bf16 -> f32 accumulate.
typedef __attribute__((ext_vector_type(16))) __bf16 v16bf;
typedef __attribute__((ext_vector_type(8)))  __bf16 v8bf;
typedef __attribute__((ext_vector_type(8)))  float  v8f;

#define INPUT_DIM  1024
#define OUTPUT_DIM 1024
#define NDEG 9          // D+1
#define KPAD 16         // degree slots padded to 16 (d=9..15 are zeros)
#define BM 128          // C-tile rows per workgroup
#define BN 128          // C-tile cols per workgroup
#define IPC 4           // input features per K-chunk
#define KC  (IPC*KPAD)  // 64 K elements per chunk = 2 WMMA K-steps
#define LDS_STRIDE 72   // 64 K-elems + 8 pad (bf16); row = 144B, 16B-aligned

// Assemble one 16x16x32 bf16 fragment (A 16xK, or B KxN read column-wise)
// from a row-major [outer][LDS_STRIDE] bf16 LDS tile, per the ISA 16-bit
// operand striping: lanes 0-15 hold K=0..7 & 16..23, lanes 16-31 hold
// K=8..15 & 24..31; outer index = lane%16.
__device__ __forceinline__ v16bf frag_from_lds(const __bf16* tile, int outer_base,
                                               int koff, int lane) {
  const int r  = outer_base + (lane & 15);
  const int kh = (lane >> 4) << 3;                 // 0 or 8
  const __bf16* p = tile + r * LDS_STRIDE + koff + kh;
  v8bf lo = *(const v8bf*)(p);                     // K = kh .. kh+7
  v8bf hi = *(const v8bf*)(p + 16);                // K = kh+16 .. kh+23
  return __builtin_shufflevector(lo, hi, 0,1,2,3,4,5,6,7,8,9,10,11,12,13,14,15);
}

// Branch-free tanh: 1 - 2/(1+e^{2x}); saturates to +-1 at the extremes,
// then clipped to +-0.999 like the reference. One v_exp_f32 + one v_rcp_f32,
// no EXEC divergence (keeps the wave WMMA-legal and cheap).
__device__ __forceinline__ float fast_tanh(float x) {
  float e = __expf(2.0f * x);
  float t = 1.0f - 2.0f * __builtin_amdgcn_rcpf(1.0f + e);
  return fminf(fmaxf(t, -0.999f), 0.999f);
}

__global__ __launch_bounds__(256)
void cheby_kan_wmma(const float* __restrict__ x,
                    const float* __restrict__ coeffs,
                    float* __restrict__ out) {
  __shared__ __align__(16) __bf16 ldsA[BM * LDS_STRIDE]; // basis tile [m][k]
  __shared__ __align__(16) __bf16 ldsB[BN * LDS_STRIDE]; // coeff tile [o][k]

  const int tid  = threadIdx.x;
  const int lane = tid & 31;
  const int wave = tid >> 5;        // 0..7
  const int wm   = wave >> 2;       // 0..1  -> 64-row strip
  const int wn   = wave & 3;        // 0..3  -> 32-col strip
  const int row0 = blockIdx.x * BM;
  const int col0 = blockIdx.y * BN;

  v8f acc[4][2];
  #pragma unroll
  for (int a = 0; a < 4; ++a)
    #pragma unroll
    for (int b = 0; b < 2; ++b)
      acc[a][b] = v8f{0.f,0.f,0.f,0.f,0.f,0.f,0.f,0.f};

  for (int i0 = 0; i0 < INPUT_DIM; i0 += IPC) {
    __syncthreads();   // previous chunk's fragment reads done before overwrite

    // ---- Stage A & B tiles: 512 (row, i_local) tasks over 256 threads
    #pragma unroll
    for (int task = 0; task < 2; ++task) {
      const int tt  = tid + task * 256;
      const int tm  = tt & 127;          // staged row (A) / col (B)
      const int til = tt >> 7;           // i_local 0..3
      const int i   = i0 + til;

      // A: Chebyshev basis via recurrence (== cos(d*arccos(t)))
      {
        const float t = fast_tanh(x[(size_t)(row0 + tm) * INPUT_DIM + i]);
        float T[NDEG];
        T[0] = 1.f; T[1] = t;
        const float t2 = 2.f * t;
        #pragma unroll
        for (int d = 2; d < NDEG; ++d) T[d] = t2 * T[d-1] - T[d-2];
        v8bf lo = {(__bf16)T[0], (__bf16)T[1], (__bf16)T[2], (__bf16)T[3],
                   (__bf16)T[4], (__bf16)T[5], (__bf16)T[6], (__bf16)T[7]};
        v8bf hi = {(__bf16)T[8], (__bf16)0.f, (__bf16)0.f, (__bf16)0.f,
                   (__bf16)0.f,  (__bf16)0.f, (__bf16)0.f, (__bf16)0.f};
        __bf16* dst = ldsA + tm * LDS_STRIDE + til * KPAD;
        *(v8bf*)dst       = lo;
        *(v8bf*)(dst + 8) = hi;
      }

      // B: coeffs[i][o][d] f32 -> bf16, layout [o][i_local*16+d]
      {
        const float* g = coeffs + ((size_t)i * OUTPUT_DIM + (col0 + tm)) * NDEG;
        float c[NDEG];
        #pragma unroll
        for (int d = 0; d < NDEG; ++d) c[d] = g[d];
        if (i0 + IPC < INPUT_DIM)   // uniform; lowers to global_prefetch_b8
          __builtin_prefetch(coeffs + ((size_t)(i + IPC) * OUTPUT_DIM + (col0 + tm)) * NDEG, 0, 0);
        v8bf lo = {(__bf16)c[0], (__bf16)c[1], (__bf16)c[2], (__bf16)c[3],
                   (__bf16)c[4], (__bf16)c[5], (__bf16)c[6], (__bf16)c[7]};
        v8bf hi = {(__bf16)c[8], (__bf16)0.f, (__bf16)0.f, (__bf16)0.f,
                   (__bf16)0.f,  (__bf16)0.f, (__bf16)0.f, (__bf16)0.f};
        __bf16* dst = ldsB + tm * LDS_STRIDE + til * KPAD;
        *(v8bf*)dst       = lo;
        *(v8bf*)(dst + 8) = hi;
      }
    }

    __syncthreads();

    // ---- WMMA: 2 K-steps x 4 M-frags x 2 N-frags = 16 WMMA per wave
    #pragma unroll
    for (int kc = 0; kc < 2; ++kc) {
      const int koff = kc * 32;
      v16bf bfrag[2];
      #pragma unroll
      for (int nf = 0; nf < 2; ++nf)
        bfrag[nf] = frag_from_lds(ldsB, wn * 32 + nf * 16, koff, lane);
      #pragma unroll
      for (int mf = 0; mf < 4; ++mf) {
        v16bf afrag = frag_from_lds(ldsA, wm * 64 + mf * 16, koff, lane);
        #pragma unroll
        for (int nf = 0; nf < 2; ++nf)
          acc[mf][nf] = __builtin_amdgcn_wmma_f32_16x16x32_bf16(
              false, afrag, false, bfrag[nf], (short)0, acc[mf][nf], false, false);
      }
    }
  }

  // ---- Epilogue: f32 C fragment layout -> global
  // VGPR r: lanes 0-15 => M=r, N=lane; lanes 16-31 => M=8+r, N=lane-16.
  const int mrow = (lane >> 4) << 3;   // 0 or 8
  const int ncol = lane & 15;
  #pragma unroll
  for (int mf = 0; mf < 4; ++mf) {
    #pragma unroll
    for (int nf = 0; nf < 2; ++nf) {
      #pragma unroll
      for (int r = 0; r < 8; ++r) {
        const int grow = row0 + wm * 64 + mf * 16 + mrow + r;
        const int gcol = col0 + wn * 32 + nf * 16 + ncol;
        out[(size_t)grow * OUTPUT_DIM + gcol] = acc[mf][nf][r];
      }
    }
  }
}

extern "C" void kernel_launch(void* const* d_in, const int* in_sizes, int n_in,
                              void* d_out, int out_size, void* d_ws, size_t ws_size,
                              hipStream_t stream) {
  const float* x      = (const float*)d_in[0];   // [B, 1024] f32
  const float* coeffs = (const float*)d_in[1];   // [1024, 1024, 9] f32
  float* out          = (float*)d_out;           // [B, 1024] f32

  const int batch = in_sizes[0] / INPUT_DIM;     // 8192
  dim3 grid(batch / BM, OUTPUT_DIM / BN);        // (64, 8)
  cheby_kan_wmma<<<grid, 256, 0, stream>>>(x, coeffs, out);
}
